// ISMLBlock_61761629716818
// MI455X (gfx1250) — matrix operand order repaired and checked by reading, not compiled, not run on hardware
//
#include <hip/hip_runtime.h>
#include <math.h>

#define BB 32
#define SS 128
#define HH 256
#define H2 512
#define G4 1024
#define MROWS 4096      // B*S
#define XPAD 544        // K padded to multiple of 32 (covers 512/516/520)
#define STMP_LD 524
#define NITER 3
#define NLAYER 4
#define DOUT 128

typedef _Float16 v16h __attribute__((ext_vector_type(16)));
typedef float    v8f  __attribute__((ext_vector_type(8)));
typedef _Float16 v8h  __attribute__((ext_vector_type(8)));

// Pack two contiguous float4 runs into 8 f16 elements of a fragment,
// per ISA A VGPR layout (K-pairs are contiguous per lane).
__device__ inline void pack8(v16h& v, int base, float4 a, float4 b)
{
    v[base + 0] = (_Float16)a.x;  v[base + 1] = (_Float16)a.y;
    v[base + 2] = (_Float16)a.z;  v[base + 3] = (_Float16)a.w;
    v[base + 4] = (_Float16)b.x;  v[base + 5] = (_Float16)b.y;
    v[base + 6] = (_Float16)b.z;  v[base + 7] = (_Float16)b.w;
}

__device__ inline float4 lrelu4(float4 v)
{
    v.x = v.x > 0.f ? v.x : 0.01f * v.x;
    v.y = v.y > 0.f ? v.y : 0.01f * v.y;
    v.z = v.z > 0.f ? v.z : 0.01f * v.z;
    v.w = v.w > 0.f ? v.w : 0.01f * v.w;
    return v;
}

// ---------------------------------------------------------------------------
// GEMM: Y[m,n] = act(X)[m,:] . Wh[n,:] + b0[n] + b1[n]
// X f32 [.., LDX] zero-padded; Wh f16 [Nt*16, LDW] pre-converted/padded.
// One wave computes a 64x16 block. Per k-step: load B + 4 A fragments into
// distinct registers (one clause), then 4 back-to-back WMMAs.
// ---------------------------------------------------------------------------
template <int ACT, int LDX, int LDW>
__global__ __launch_bounds__(256) void gemm_wmma_f16(
    const float* __restrict__ X,
    const _Float16* __restrict__ Wh,
    const float* __restrict__ b0, const float* __restrict__ b1,
    float* __restrict__ Y, int ldy,
    int Mt4, int Nt, int KT)
{
    int lane = threadIdx.x & 31;
    int wv   = (blockIdx.x * blockDim.x + threadIdx.x) >> 5;
    if (wv >= Mt4 * Nt) return;                       // wave-uniform
    int mt = wv / Nt, nt = wv % Nt;
    int m0 = mt << 6, n0 = nt << 4;
    int l15 = lane & 15;
    int hi  = lane >> 4;

    const _Float16* wp  = Wh + (size_t)(n0 + l15) * LDW + (hi << 3);
    const float*    xp0 = X  + (size_t)(m0 + l15) * LDX + (hi << 3);

    v8f acc[4] = {};
    for (int kt = 0; kt < KT; ++kt) {
        __builtin_prefetch(wp + 128, 0, 1);
        v16h bf = __builtin_shufflevector(*(const v8h*)wp, *(const v8h*)(wp + 16), 0,1,2,3,4,5,6,7,8,9,10,11,12,13,14,15); // two 8-f16 runs
        v16h afs[4];
#pragma unroll
        for (int i = 0; i < 4; ++i) {
            const float* xi = xp0 + i * (16 * LDX);   // constant offsets
            float4 a0 = *(const float4*)(xi + 0);
            float4 a1 = *(const float4*)(xi + 4);
            float4 a2 = *(const float4*)(xi + 16);
            float4 a3 = *(const float4*)(xi + 20);
            if (ACT) { a0 = lrelu4(a0); a1 = lrelu4(a1); a2 = lrelu4(a2); a3 = lrelu4(a3); }
            pack8(afs[i], 0, a0, a1);
            pack8(afs[i], 8, a2, a3);
        }
#pragma unroll
        for (int i = 0; i < 4; ++i)
            acc[i] = __builtin_amdgcn_wmma_f32_16x16x32_f16(
                false, afs[i], false, bf, (short)0, acc[i], false, false);
        wp += 32;
        xp0 += 32;
    }
    float bias = 0.f;
    if (b0) bias += b0[n0 + l15];
    if (b1) bias += b1[n0 + l15];
#pragma unroll
    for (int i = 0; i < 4; ++i) {
        int mb = m0 + (i << 4) + (hi << 3);
#pragma unroll
        for (int r = 0; r < 8; ++r)
            Y[(size_t)(mb + r) * ldy + n0 + l15] = acc[i][r] + bias;
    }
}

// ---------------------------------------------------------------------------
// Persistent bidirectional LSTM scan. grid.x == 2 (0 = fwd, 1 = bwd).
// Whh pre-converted to f16. Per step: stage xp(:,t,:) into LDS, then
// gates += h @ Whh^T in two passes of 8 n-tiles: per k-step load A (LDS) and
// all 8 B fragments (one clause, constant offsets), then 8 WMMAs back-to-back.
// ---------------------------------------------------------------------------
__global__ __launch_bounds__(256) void lstm_scan(
    const float* __restrict__ xpF, const float* __restrict__ xpB,
    const _Float16* __restrict__ whhF, const _Float16* __restrict__ whhB,
    float* __restrict__ out, int oldld)
{
    const float*    xp  = blockIdx.x ? xpB  : xpF;
    const _Float16* whh = blockIdx.x ? whhB : whhF;
    const bool  rev  = blockIdx.x != 0;
    const int   colo = blockIdx.x * HH;

    extern __shared__ float sm[];
    float* hb = sm;                 // [32][256]
    float* cb = sm + BB * HH;       // [32][256]
    float* gb = sm + 2 * BB * HH;   // [32][1024] gates (pre-loaded with xp(t))

    int tid  = threadIdx.x;
    int lane = tid & 31;
    int wid  = tid >> 5;            // 0..7
    int l15  = lane & 15;
    int hi   = lane >> 4;
    int m0   = (wid & 1) << 4;      // batch-tile base (0 or 16)
    int ntb  = (wid >> 1) << 4;     // first of 16 n-tiles for this wave

    const float4* xp4 = (const float4*)xp;
    float4*       gb4 = (float4*)gb;

    for (int e = tid; e < BB * HH; e += 256) { hb[e] = 0.f; cb[e] = 0.f; }
    __syncthreads();

    for (int tt = 0; tt < SS; ++tt) {
        int t = rev ? (SS - 1 - tt) : tt;

        // stage xp(:, t, :) -> gb, coalesced b128
#pragma unroll 4
        for (int b = 0; b < BB; ++b)
            gb4[b * 256 + tid] = xp4[(size_t)(b * SS + t) * 256 + tid];
        __syncthreads();

        // gates = gb + h @ Whh^T : two passes of 8 n-tiles
#pragma unroll 1
        for (int p = 0; p < 2; ++p) {
            int nb = ntb + (p << 3);
            const _Float16* wb =
                whh + (size_t)((nb << 4) + l15) * HH + (hi << 3);
            v8f acc[8];
#pragma unroll
            for (int i = 0; i < 8; ++i) {
                int g = ((nb + i) << 4) + l15;
#pragma unroll
                for (int r = 0; r < 8; ++r)
                    acc[i][r] = gb[(m0 + (hi << 3) + r) * G4 + g];
            }
#pragma unroll 1
            for (int k0 = 0; k0 < HH; k0 += 32) {
                const float* hr = &hb[(m0 + l15) * HH + k0 + (hi << 3)];
                float4 h0 = *(const float4*)(hr + 0);
                float4 h1 = *(const float4*)(hr + 4);
                float4 h2 = *(const float4*)(hr + 16);
                float4 h3 = *(const float4*)(hr + 20);
                v16h af;
                pack8(af, 0, h0, h1);
                pack8(af, 8, h2, h3);
                v16h bfs[8];
#pragma unroll
                for (int i = 0; i < 8; ++i)          // constant offsets i*16*HH
                    bfs[i] = __builtin_shufflevector(*(const v8h*)(wb + i * (16 * HH) + k0), *(const v8h*)(wb + i * (16 * HH) + k0 + 16), 0,1,2,3,4,5,6,7,8,9,10,11,12,13,14,15);
#pragma unroll
                for (int i = 0; i < 8; ++i)
                    acc[i] = __builtin_amdgcn_wmma_f32_16x16x32_f16(
                        false, af, false, bfs[i], (short)0, acc[i], false, false);
            }
#pragma unroll
            for (int i = 0; i < 8; ++i) {
                int g = ((nb + i) << 4) + l15;
#pragma unroll
                for (int r = 0; r < 8; ++r)
                    gb[(m0 + (hi << 3) + r) * G4 + g] = acc[i][r];
            }
        }
        __syncthreads();

        // LSTM cell (gate order i, f, g, o)
        for (int e = tid; e < BB * HH; e += 256) {
            int b = e >> 8, j = e & 255;
            float ig = gb[b * G4 + j];
            float fg = gb[b * G4 + HH + j];
            float gg = gb[b * G4 + 2 * HH + j];
            float og = gb[b * G4 + 3 * HH + j];
            ig = 1.f / (1.f + __expf(-ig));
            fg = 1.f / (1.f + __expf(-fg));
            og = 1.f / (1.f + __expf(-og));
            gg = tanhf(gg);
            float c = fg * cb[e] + ig * gg;
            float h = og * tanhf(c);
            cb[e] = c;
            hb[e] = h;
            out[(size_t)(b * SS + t) * oldld + colo + j] = h;
        }
        __syncthreads();
    }
}

// ---------------------------------------------------------------------------
__global__ void copy_pad(const float* __restrict__ src, int lds, int width,
                         float* __restrict__ dst, int ldd, int dwidth, int rows)
{
    int idx = blockIdx.x * blockDim.x + threadIdx.x;
    if (idx >= rows * dwidth) return;
    int m = idx / dwidth, c = idx % dwidth;
    dst[(size_t)m * ldd + c] = (c < width) ? src[(size_t)m * lds + c] : 0.f;
}

// f32 -> f16 weight conversion with zero padding to dwidth columns.
__global__ void cvt_pad_f16(const float* __restrict__ src, int lds, int width,
                            _Float16* __restrict__ dst, int ldd, int dwidth, int rows)
{
    int idx = blockIdx.x * blockDim.x + threadIdx.x;
    if (idx >= rows * dwidth) return;
    int m = idx / dwidth, c = idx % dwidth;
    dst[(size_t)m * ldd + c] =
        (c < width) ? (_Float16)src[(size_t)m * lds + c] : (_Float16)0.f;
}

// ---------------------------------------------------------------------------
// y = softmax(X @ W^T + b), W [2,512]; result also appended into s_tmp.
// ---------------------------------------------------------------------------
__global__ void fc2_softmax(const float* __restrict__ X, int ldx,
                            const float* __restrict__ W, const float* __restrict__ b,
                            float* __restrict__ y, float* __restrict__ stmp_col, int rows)
{
    int m = blockIdx.x * blockDim.x + threadIdx.x;
    if (m >= rows) return;
    const float* x = X + (size_t)m * ldx;
    float z0 = b[0], z1 = b[1];
    for (int k = 0; k < H2; ++k) {
        float xv = x[k];
        z0 += xv * W[k];
        z1 += xv * W[H2 + k];
    }
    float mx = fmaxf(z0, z1);
    float e0 = __expf(z0 - mx), e1 = __expf(z1 - mx);
    float inv = 1.f / (e0 + e1);
    float p0 = e0 * inv, p1 = e1 * inv;
    y[2 * m]     = p0;
    y[2 * m + 1] = p1;
    stmp_col[(size_t)m * STMP_LD]     = p0;
    stmp_col[(size_t)m * STMP_LD + 1] = p1;
}

// ---------------------------------------------------------------------------
extern "C" void kernel_launch(void* const* d_in, const int* in_sizes, int n_in,
                              void* d_out, int out_size, void* d_ws, size_t ws_size,
                              hipStream_t stream)
{
    (void)in_sizes; (void)n_in; (void)out_size;
    const float* s1 = (const float*)d_in[0];
    auto P = [&](int i) { return (const float*)d_in[i]; };

    // Nested pytrees flatten lists in order, dicts by sorted key:
    // per direction {Whh, Wih, bhh, bih}; fc dicts {W, b}.
    const int base_e  = 3;
    const int base_c  = 3 + 96;
    const int base_fe = 3 + 192;
    const int base_fc = 3 + 198;
    const int i_cmlW = 3 + 204, i_cmlb = 3 + 205, i_emlW = 3 + 206, i_emlb = 3 + 207;
    auto widx = [&](int base, int n, int layer, int dir, int which) {
        return base + (((n * NLAYER + layer) * 2 + dir) * 4) + which;
    }; // which: 0=Whh 1=Wih 2=bhh 3=bih

    // Workspace layout
    float* ws   = (float*)d_ws;
    float* xA   = ws;
    float* xB   = xA  + (size_t)MROWS * XPAD;
    float* xpF  = xB  + (size_t)MROWS * XPAD;
    float* xpB  = xpF + (size_t)MROWS * G4;
    float* eout = xpB + (size_t)MROWS * G4;
    float* cout = eout + (size_t)MROWS * H2;
    _Float16* wihH0 = (_Float16*)(cout + (size_t)MROWS * H2);
    _Float16* wihH1 = wihH0 + (size_t)G4 * XPAD;
    _Float16* whhH0 = wihH1 + (size_t)G4 * XPAD;
    _Float16* whhH1 = whhH0 + (size_t)G4 * HH;
    size_t need = (size_t)MROWS * (XPAD * 2 + G4 * 2 + H2 * 2) * sizeof(float)
                + ((size_t)G4 * XPAD * 2 + (size_t)G4 * HH * 2) * sizeof(_Float16);
    if (ws_size < need) return;

    // Output layout (flat, return order)
    float* out  = (float*)d_out;
    float* ye[3] = { out,          out + 8192,  out + 16384 };
    float* yc[3] = { out + 24576,  out + 32768, out + 40960 };
    float* stmp = out + 49152;                        // [4096, 524]
    float* cml  = stmp + (size_t)MROWS * STMP_LD;     // [4096, 128]
    float* eml  = cml  + (size_t)MROWS * DOUT;        // [4096, 128]

    const int TPB = 256;
    const size_t SCAN_LDS = (size_t)(2 * BB * HH + BB * G4) * sizeof(float); // 192 KB
    auto nblk = [&](int total) { return (total + TPB - 1) / TPB; };

    // s1 -> s_tmp cols [0,512)
    copy_pad<<<nblk(MROWS * H2), TPB, 0, stream>>>(
        s1, H2, H2, stmp, STMP_LD, H2, MROWS);

    int w = H2;
    for (int n = 0; n < NITER; ++n) {
        for (int ec = 0; ec < 2; ++ec) {            // 0 = e-LSTM, 1 = c-LSTM
            int    base = ec ? base_c : base_e;
            float* fin  = ec ? cout   : eout;

            // pack s_tmp (width w) -> xA, zero-padded to 544
            copy_pad<<<nblk(MROWS * XPAD), TPB, 0, stream>>>(
                stmp, STMP_LD, w, xA, XPAD, XPAD, MROWS);

            const float* xin = xA;
            float*       xnx = xB;
            for (int layer = 0; layer < NLAYER; ++layer) {
                int K  = (layer == 0) ? w : H2;
                int KT = (K + 31) >> 5;
                const float* whh0 = P(widx(base, n, layer, 0, 0));
                const float* wih0 = P(widx(base, n, layer, 0, 1));
                const float* bhh0 = P(widx(base, n, layer, 0, 2));
                const float* bih0 = P(widx(base, n, layer, 0, 3));
                const float* whh1 = P(widx(base, n, layer, 1, 0));
                const float* wih1 = P(widx(base, n, layer, 1, 1));
                const float* bhh1 = P(widx(base, n, layer, 1, 2));
                const float* bih1 = P(widx(base, n, layer, 1, 3));

                // weights -> f16 (Wih zero-padded to XPAD columns)
                cvt_pad_f16<<<nblk(G4 * XPAD), TPB, 0, stream>>>(
                    wih0, K, K, wihH0, XPAD, XPAD, G4);
                cvt_pad_f16<<<nblk(G4 * XPAD), TPB, 0, stream>>>(
                    wih1, K, K, wihH1, XPAD, XPAD, G4);
                cvt_pad_f16<<<nblk(G4 * HH), TPB, 0, stream>>>(
                    whh0, HH, HH, whhH0, HH, HH, G4);
                cvt_pad_f16<<<nblk(G4 * HH), TPB, 0, stream>>>(
                    whh1, HH, HH, whhH1, HH, HH, G4);

                int waves  = (MROWS / 64) * (G4 / 16);
                int blocks = nblk(waves * 32);
                gemm_wmma_f16<0, XPAD, XPAD><<<blocks, TPB, 0, stream>>>(
                    xin, wihH0, bih0, bhh0, xpF, G4, MROWS / 64, G4 / 16, KT);
                gemm_wmma_f16<0, XPAD, XPAD><<<blocks, TPB, 0, stream>>>(
                    xin, wihH1, bih1, bhh1, xpB, G4, MROWS / 64, G4 / 16, KT);

                float* obuf  = (layer == NLAYER - 1) ? fin : xnx;
                int    oldld = (layer == NLAYER - 1) ? H2  : XPAD;
                lstm_scan<<<2, TPB, SCAN_LDS, stream>>>(
                    xpF, xpB, whhH0, whhH1, obuf, oldld);

                if (layer < NLAYER - 1) {
                    float* t = (float*)xin;
                    xin = xnx;
                    xnx = t;
                }
            }
            // softmax head + append into s_tmp
            const float* fw = P((ec ? base_fc : base_fe) + 2 * n);
            const float* fb = P((ec ? base_fc : base_fe) + 2 * n + 1);
            float* yout = ec ? yc[n] : ye[n];
            int coff = w + (ec ? 2 : 0);
            fc2_softmax<<<nblk(MROWS), TPB, 0, stream>>>(
                fin, H2, fw, fb, yout, stmp + coff, MROWS);
        }
        w += 4;
    }

    // final scores: cml = lrelu(e_out) @ cmlW^T + b ; eml = lrelu(c_out) @ emlW^T + b
    {
        cvt_pad_f16<<<nblk(DOUT * H2), TPB, 0, stream>>>(
            P(i_cmlW), H2, H2, wihH0, H2, H2, DOUT);
        cvt_pad_f16<<<nblk(DOUT * H2), TPB, 0, stream>>>(
            P(i_emlW), H2, H2, wihH1, H2, H2, DOUT);
        int waves  = (MROWS / 64) * (DOUT / 16);
        int blocks = nblk(waves * 32);
        gemm_wmma_f16<1, H2, H2><<<blocks, TPB, 0, stream>>>(
            eout, wihH0, P(i_cmlb), nullptr, cml, DOUT, MROWS / 64, DOUT / 16, H2 / 32);
        gemm_wmma_f16<1, H2, H2><<<blocks, TPB, 0, stream>>>(
            cout, wihH1, P(i_emlb), nullptr, eml, DOUT, MROWS / 64, DOUT / 16, H2 / 32);
    }
}
